// BasicForward_46273977647338
// MI455X (gfx1250) — matrix operand order repaired and checked by reading, not compiled, not run on hardware
//
#include <hip/hip_runtime.h>

// CDNA5 / gfx1250 fused type-2 NUFFT:
//   kspace[b,m] = sum_y ey[m,y] * ( sum_x ex[m,x] * img[b,x,y] )
//
// Fast path: fp32 GEMM emulated with split-f16 triple products on
// V_WMMA_F32_16X16X32_F16 (a*b ~= ah*bh + ah*bl + al*bh, fp32 accumulate)
// -> ~2.7x fewer matrix instructions than the fp32 K=4 WMMA path.
// Fallback path (ws too small): direct V_WMMA_F32_16X16X4_F32.

typedef _Float16 v16h __attribute__((ext_vector_type(16)));
typedef float    v2f  __attribute__((ext_vector_type(2)));
typedef float    v8f  __attribute__((ext_vector_type(8)));

#define B_     2
#define NX_    256
#define NY_    256
#define M_     16384
#define NCHUNK (NX_ / 32)   // 8 K-chunks of 32
#define KSTEPS (NX_ / 4)    // 64 K-steps of 4 (fallback)
#define YTILES (NY_ / 16)   // 16 y-tiles

// d_ws layout: [hi: B_*NCHUNK*NY_*32 halfs][lo: same]  = 512 KB total
#define SPLIT_HALFS ((size_t)B_ * NCHUNK * NY_ * 32)

// ---------------------------------------------------------------------------
// Prep: split fp32 image into f16 hi/lo, stored x-major per (b, chunk, y) so
// the main kernel's B operand is one aligned 32B v16h load per lane.
// half index = ((b*NCHUNK + c)*NY_ + y)*32 + (x - 32c)
// ---------------------------------------------------------------------------
__global__ __launch_bounds__(256)
void img_split_kernel(const float* __restrict__ img,
                      _Float16* __restrict__ hi,
                      _Float16* __restrict__ lo)
{
    const int t = blockIdx.x * 256 + threadIdx.x;   // 0 .. B_*NCHUNK*NY_-1
    const int y = t & (NY_ - 1);
    const int c = (t >> 8) & (NCHUNK - 1);
    const int b = t >> 11;

    const float* __restrict__ src =
        img + ((size_t)b * NX_ + (size_t)c * 32) * NY_ + y;

    v16h h0, h1, l0, l1;
    #pragma unroll
    for (int x = 0; x < 16; ++x) {
        float v = src[(size_t)x * NY_];
        _Float16 hh = (_Float16)v;
        h0[x] = hh;
        l0[x] = (_Float16)(v - (float)hh);
    }
    #pragma unroll
    for (int x = 0; x < 16; ++x) {
        float v = src[(size_t)(x + 16) * NY_];
        _Float16 hh = (_Float16)v;
        h1[x] = hh;
        l1[x] = (_Float16)(v - (float)hh);
    }

    const size_t base = (size_t)t * 32;             // halfs; 64B aligned
    *(v16h*)(hi + base)      = h0;
    *(v16h*)(hi + base + 16) = h1;
    *(v16h*)(lo + base)      = l0;
    *(v16h*)(lo + base + 16) = l1;
}

// ---------------------------------------------------------------------------
// Fast main kernel: one wave32 per 16-wide m-tile, split-f16 WMMA.
// ---------------------------------------------------------------------------
__global__ __launch_bounds__(32)
void nufft2_wmma_f16x3_kernel(const float* __restrict__ traj,
                              const _Float16* __restrict__ bhi,
                              const _Float16* __restrict__ blo,
                              float* __restrict__ out)
{
    // A operands (split f16 exponentials), per-lane 32B blocks:
    // [chunk][mat: re_hi,re_lo,im_hi,im_lo][lane][16 halfs]  = 32 KB
    __shared__ __align__(32) _Float16 ldsA[NCHUNK * 4 * 32 * 16];
    v16h* __restrict__ ldsAv = (v16h*)ldsA;

    const int lane = threadIdx.x & 31;
    const int half = lane >> 4;
    const int l16  = lane & 15;

    const int blk = blockIdx.x;
    const int b   = blk >> 10;             // / (M_/16)
    const int mt  = (blk & 1023) << 4;     // m-tile base

    const float* __restrict__ kxp = traj;
    const float* __restrict__ kyp = traj + M_;

    // ---- Phase 1: build split-f16 A tiles in LDS (once per m-tile) ----
    // 16-bit A 16x32 layout (contiguous, matches B): lanes 0-15 (M=lane): K = 0..15;
    // lanes 16-31 (M=lane-16): K = 16..31.
    const float kx = kxp[mt + l16];
    for (int c = 0; c < NCHUNK; ++c) {
        const int kbase = 32 * c + (half ? 16 : 0);
        v16h rh, rl, ih, il;
        #pragma unroll
        for (int j = 0; j < 16; ++j) {
            const int k  = kbase + j;
            const float xp  = (float)(k - NX_ / 2);
            const float ang = kx * xp;
            const float cc  = __cosf(ang);
            const float ms  = -__sinf(ang);           // ex = cos - i sin
            _Float16 a = (_Float16)cc;
            rh[j] = a;
            rl[j] = (_Float16)(cc - (float)a);
            _Float16 bq = (_Float16)ms;
            ih[j] = bq;
            il[j] = (_Float16)(ms - (float)bq);
        }
        ldsAv[(c * 4 + 0) * 32 + lane] = rh;
        ldsAv[(c * 4 + 1) * 32 + lane] = rl;
        ldsAv[(c * 4 + 2) * 32 + lane] = ih;
        ldsAv[(c * 4 + 3) * 32 + lane] = il;
    }
    __syncthreads();

    // ky per accumulator row r: C layout puts M=r in lanes 0-15, M=r+8 in 16-31
    float kysel[8];
    #pragma unroll
    for (int r = 0; r < 8; ++r)
        kysel[r] = kyp[mt + r + (half ? 8 : 0)];

    float accre[8], accim[8];
    #pragma unroll
    for (int r = 0; r < 8; ++r) { accre[r] = 0.0f; accim[r] = 0.0f; }

    // B base (halfs): ((b*NCHUNK + c)*NY_ + y)*32 + half*16
    const size_t bb = (size_t)b * NCHUNK;

    // ---- Main loop: y-tile pairs, 8 K-chunks each ----
    for (int yp = 0; yp < YTILES / 2; ++yp) {
        const int y0 = (yp << 5) + l16;        // first tile's y
        const int y1 = y0 + 16;                // second tile's y

        v8f cre0 = {0.f,0.f,0.f,0.f,0.f,0.f,0.f,0.f};
        v8f cim0 = {0.f,0.f,0.f,0.f,0.f,0.f,0.f,0.f};
        v8f cre1 = {0.f,0.f,0.f,0.f,0.f,0.f,0.f,0.f};
        v8f cim1 = {0.f,0.f,0.f,0.f,0.f,0.f,0.f,0.f};

        #pragma unroll 2
        for (int c = 0; c < NCHUNK; ++c) {
            const v16h arh = ldsAv[(c * 4 + 0) * 32 + lane];
            const v16h arl = ldsAv[(c * 4 + 1) * 32 + lane];
            const v16h aih = ldsAv[(c * 4 + 2) * 32 + lane];
            const v16h ail = ldsAv[(c * 4 + 3) * 32 + lane];

            const size_t s0 = (((bb + c) * NY_) + y0) * 32 + (half ? 16 : 0);
            const size_t s1 = (((bb + c) * NY_) + y1) * 32 + (half ? 16 : 0);
            const v16h bh0 = *(const v16h*)(bhi + s0);
            const v16h bl0 = *(const v16h*)(blo + s0);
            const v16h bh1 = *(const v16h*)(bhi + s1);
            const v16h bl1 = *(const v16h*)(blo + s1);

            // (neg_a, A, neg_b, B, c_mod, C, reuse_a, reuse_b)
            cre0 = __builtin_amdgcn_wmma_f32_16x16x32_f16(false, arh, false, bh0, (short)0, cre0, false, false);
            cre0 = __builtin_amdgcn_wmma_f32_16x16x32_f16(false, arh, false, bl0, (short)0, cre0, false, false);
            cre0 = __builtin_amdgcn_wmma_f32_16x16x32_f16(false, arl, false, bh0, (short)0, cre0, false, false);
            cim0 = __builtin_amdgcn_wmma_f32_16x16x32_f16(false, aih, false, bh0, (short)0, cim0, false, false);
            cim0 = __builtin_amdgcn_wmma_f32_16x16x32_f16(false, aih, false, bl0, (short)0, cim0, false, false);
            cim0 = __builtin_amdgcn_wmma_f32_16x16x32_f16(false, ail, false, bh0, (short)0, cim0, false, false);

            cre1 = __builtin_amdgcn_wmma_f32_16x16x32_f16(false, arh, false, bh1, (short)0, cre1, false, false);
            cre1 = __builtin_amdgcn_wmma_f32_16x16x32_f16(false, arh, false, bl1, (short)0, cre1, false, false);
            cre1 = __builtin_amdgcn_wmma_f32_16x16x32_f16(false, arl, false, bh1, (short)0, cre1, false, false);
            cim1 = __builtin_amdgcn_wmma_f32_16x16x32_f16(false, aih, false, bh1, (short)0, cim1, false, false);
            cim1 = __builtin_amdgcn_wmma_f32_16x16x32_f16(false, aih, false, bl1, (short)0, cim1, false, false);
            cim1 = __builtin_amdgcn_wmma_f32_16x16x32_f16(false, ail, false, bh1, (short)0, cim1, false, false);
        }

        // Fused second contraction on both tiles of the pair
        const float yp0 = (float)(y0 - NY_ / 2);
        const float yp1 = (float)(y1 - NY_ / 2);
        #pragma unroll
        for (int r = 0; r < 8; ++r) {
            float ang = kysel[r] * yp0;
            float cc = __cosf(ang), ss = __sinf(ang);
            float tr = cre0[r], ti = cim0[r];
            accre[r] += tr * cc + ti * ss;
            accim[r] += ti * cc - tr * ss;

            ang = kysel[r] * yp1;
            cc = __cosf(ang); ss = __sinf(ang);
            tr = cre1[r]; ti = cim1[r];
            accre[r] += tr * cc + ti * ss;
            accim[r] += ti * cc - tr * ss;
        }
    }

    // ---- Cross-lane reduction over the 16 y-lanes of each half ----
    #pragma unroll
    for (int r = 0; r < 8; ++r) {
        float re = accre[r];
        float im = accim[r];
        #pragma unroll
        for (int off = 1; off < 16; off <<= 1) {
            re += __shfl_xor(re, off, 32);
            im += __shfl_xor(im, off, 32);
        }
        if (l16 == 0) {
            const int m = mt + r + (half ? 8 : 0);
            out[((size_t)b * M_ + m) * 2 + 0] = re;
            out[((size_t)b * M_ + m) * 2 + 1] = im;
        }
    }
}

// ---------------------------------------------------------------------------
// Fallback: direct fp32 WMMA kernel (no workspace needed).
// ---------------------------------------------------------------------------
__global__ __launch_bounds__(32)
void nufft2_wmma_f32_kernel(const float* __restrict__ image,
                            const float* __restrict__ traj,
                            float* __restrict__ out)
{
    __shared__ float ldsA[2][KSTEPS][64];

    const int lane = threadIdx.x & 31;
    const int half = lane >> 4;
    const int l16  = lane & 15;

    const int blk = blockIdx.x;
    const int b   = blk >> 10;
    const int mt  = (blk & 1023) << 4;

    const float* __restrict__ imgp = image + (size_t)b * NX_ * NY_;
    const float* __restrict__ kxp  = traj;
    const float* __restrict__ kyp  = traj + M_;

    const float kx = kxp[mt + l16];
    for (int k = 0; k < KSTEPS; ++k) {
        const int xb  = 4 * k + (half ? 2 : 0);
        const float a0 = kx * (float)(xb     - NX_ / 2);
        const float a1 = kx * (float)(xb + 1 - NX_ / 2);
        ldsA[0][k][2 * lane + 0] =  __cosf(a0);
        ldsA[0][k][2 * lane + 1] =  __cosf(a1);
        ldsA[1][k][2 * lane + 0] = -__sinf(a0);
        ldsA[1][k][2 * lane + 1] = -__sinf(a1);
    }
    __syncthreads();

    float kysel[8];
    #pragma unroll
    for (int r = 0; r < 8; ++r)
        kysel[r] = kyp[mt + r + (half ? 8 : 0)];

    float accre[8], accim[8];
    #pragma unroll
    for (int r = 0; r < 8; ++r) { accre[r] = 0.0f; accim[r] = 0.0f; }

    const v2f* __restrict__ ldsAr = (const v2f*)&ldsA[0][0][0];
    const v2f* __restrict__ ldsAi = (const v2f*)&ldsA[1][0][0];
    const int kb = half ? 2 : 0;

    for (int yt = 0; yt < YTILES; ++yt) {
        const int y = (yt << 4) + l16;
        const float* __restrict__ col = imgp + y;

        v8f cre = {0.f,0.f,0.f,0.f,0.f,0.f,0.f,0.f};
        v8f cim = {0.f,0.f,0.f,0.f,0.f,0.f,0.f,0.f};

        #pragma unroll 8
        for (int k = 0; k < KSTEPS; ++k) {
            const v2f ar = ldsAr[k * 32 + lane];
            const v2f ai = ldsAi[k * 32 + lane];
            const int x = 4 * k + kb;
            v2f bv;
            bv.x = col[(size_t)x * NY_];
            bv.y = col[(size_t)(x + 1) * NY_];
            cre = __builtin_amdgcn_wmma_f32_16x16x4_f32(false, ar, false, bv, (short)0, cre, false, false);
            cim = __builtin_amdgcn_wmma_f32_16x16x4_f32(false, ai, false, bv, (short)0, cim, false, false);
        }

        const float ypf = (float)(y - NY_ / 2);
        #pragma unroll
        for (int r = 0; r < 8; ++r) {
            const float ang = kysel[r] * ypf;
            const float cc = __cosf(ang);
            const float ss = __sinf(ang);
            const float tr = cre[r];
            const float ti = cim[r];
            accre[r] += tr * cc + ti * ss;
            accim[r] += ti * cc - tr * ss;
        }
    }

    #pragma unroll
    for (int r = 0; r < 8; ++r) {
        float re = accre[r];
        float im = accim[r];
        #pragma unroll
        for (int off = 1; off < 16; off <<= 1) {
            re += __shfl_xor(re, off, 32);
            im += __shfl_xor(im, off, 32);
        }
        if (l16 == 0) {
            const int m = mt + r + (half ? 8 : 0);
            out[((size_t)b * M_ + m) * 2 + 0] = re;
            out[((size_t)b * M_ + m) * 2 + 1] = im;
        }
    }
}

extern "C" void kernel_launch(void* const* d_in, const int* in_sizes, int n_in,
                              void* d_out, int out_size, void* d_ws, size_t ws_size,
                              hipStream_t stream) {
    const float* image = (const float*)d_in[0];   // [B, NX, NY] f32
    const float* traj  = (const float*)d_in[1];   // [2, M] f32
    float* out = (float*)d_out;                   // [B, M] complex64 (interleaved)

    const dim3 grid(B_ * (M_ / 16));              // 2048 m-tiles
    const dim3 block(32);

    const size_t need = 2 * SPLIT_HALFS * sizeof(_Float16);   // 512 KB
    if (ws_size >= need) {
        _Float16* hi = (_Float16*)d_ws;
        _Float16* lo = hi + SPLIT_HALFS;
        img_split_kernel<<<dim3((B_ * NCHUNK * NY_) / 256), dim3(256), 0, stream>>>(
            image, hi, lo);
        nufft2_wmma_f16x3_kernel<<<grid, block, 0, stream>>>(traj, hi, lo, out);
    } else {
        nufft2_wmma_f32_kernel<<<grid, block, 0, stream>>>(image, traj, out);
    }
}
